// Attention_979252544393
// MI455X (gfx1250) — matrix operand order repaired and checked by reading, not compiled, not run on hardware
//
#include <hip/hip_runtime.h>
#include <stdint.h>

// ---------------------------------------------------------------------------
// Attention block for MI455X (gfx1250, wave32, WMMA).
// B=2, N=2048, D_MODEL=1024, H=16, D_HEAD=64, INNER=1024.
// All contractions via v_wmma_f32_16x16x32_bf16 (bf16 operands, f32 accum).
// Flash-attention stage double-buffers K/V blocks into LDS with
// global_load_async_to_lds_b128 + s_wait_asynccnt when available.
// ---------------------------------------------------------------------------

#define BATCH    2
#define SEQ      2048
#define DMODEL   1024
#define NHEAD    16
#define DHEAD    64
#define INNER    1024   // NHEAD * DHEAD

typedef __bf16 bf16;
typedef __attribute__((ext_vector_type(16))) __bf16 v16bf;
typedef __attribute__((ext_vector_type(8)))  __bf16 v8bf;
typedef __attribute__((ext_vector_type(8)))  float  v8f;

// Native conversion: gfx1250 has bf16 VALU (v_cvt_pk_bf16_f32), far cheaper
// than a manual RNE bit sequence (3 VALU/elem).
static __device__ inline bf16 f2bf(float f) { return (bf16)f; }

static __device__ inline v8bf cvt8(v8f f) {
    v8bf r;
    #pragma unroll
    for (int i = 0; i < 8; ++i) r[i] = f2bf(f[i]);
    return r;
}

static __device__ inline v16bf cat8(v8bf lo, v8bf hi) {
    return __builtin_shufflevector(lo, hi, 0,1,2,3,4,5,6,7,8,9,10,11,12,13,14,15);
}

static __device__ inline v8f wmma_bf16(v16bf a, v16bf b, v8f c) {
    // (neg_a, A, neg_b, B, c_mod, C, reuse_a, reuse_b)
    return __builtin_amdgcn_wmma_f32_16x16x32_bf16(false, a, false, b,
                                                   (short)0, c, false, false);
}

// A operand (16-bit, 16x32): row = lane%16; lane holds K = kb + 16*half + 0..15.
// `rowp` = pointer to this lane's row.
static __device__ inline v16bf loadA_bf16(const bf16* rowp, int kb, int half) {
    v8bf lo = *reinterpret_cast<const v8bf*>(rowp + kb + 16 * half);
    v8bf hi = *reinterpret_cast<const v8bf*>(rowp + kb + 16 * half + 8);
    return cat8(lo, hi);
}

// B operand (32x16): lane holds column n=lane%16, K = kb + 16*half + 0..15
// (contiguous when the K axis is innermost). `colp` = pointer to lane's column.
static __device__ inline v16bf loadB_bf16(const bf16* colp, int kb, int half) {
    return *reinterpret_cast<const v16bf*>(colp + kb + 16 * half);
}

static __device__ inline float rowmax16(float v) {
    #pragma unroll
    for (int off = 1; off < 16; off <<= 1) v = fmaxf(v, __shfl_xor(v, off, 32));
    return v;
}
static __device__ inline float rowsum16(float v) {
    #pragma unroll
    for (int off = 1; off < 16; off <<= 1) v += __shfl_xor(v, off, 32);
    return v;
}

// ---------------------------------------------------------------------------
// Async global -> LDS copy (gfx1250 ASYNCcnt path), with sync-LDS fallback so
// the file compiles on toolchains/host passes lacking the builtins.
// Probe-learned signature: (int4* src, int4* dst, imm offset, imm cpol).
// ---------------------------------------------------------------------------
#if __has_builtin(__builtin_amdgcn_global_load_async_to_lds_b128) && \
    __has_builtin(__builtin_amdgcn_s_wait_asynccnt)
#define HAS_ASYNC_LDS 1
typedef int v4i __attribute__((vector_size(16)));
static __device__ inline void cp16_async(bf16* lds, const bf16* g) {
    __builtin_amdgcn_global_load_async_to_lds_b128((v4i*)g, (v4i*)lds, 0, 0);
}
// Async loads complete in order: ASYNCcnt <= 16 means the 16 loads of the
// previously-issued block are done even with 16 newer loads in flight.
static __device__ inline void async_wait16() { __builtin_amdgcn_s_wait_asynccnt(16); }
static __device__ inline void async_wait0()  { __builtin_amdgcn_s_wait_asynccnt(0); }
#else
#define HAS_ASYNC_LDS 0
static __device__ inline void cp16_async(bf16* lds, const bf16* g) {
    *reinterpret_cast<v8bf*>(lds) = *reinterpret_cast<const v8bf*>(g);
}
static __device__ inline void async_wait16() {}
static __device__ inline void async_wait0()  {}
#endif

// WGP-scope prefetch (locality 3): pull into all cache levels, not just GL2.
static __device__ inline void prefetch_near(const void* p) {
    __builtin_prefetch(p, 0, 3);
}

// ---------------------------------------------------------------------------
// Kernel 0a: transpose + convert fp32 weights [K, Ncol] -> bf16 WT [Ncol, K]
// so WMMA B-operands are contiguous 32B loads. Runs once, tiny cost.
// ---------------------------------------------------------------------------
__global__ void transpose_cvt_kernel(const float* __restrict__ W,
                                     bf16* __restrict__ WT,
                                     int K, int Ncol) {
    int idx = blockIdx.x * blockDim.x + threadIdx.x;
    if (idx >= K * Ncol) return;
    int k = idx / Ncol, c = idx % Ncol;
    WT[(size_t)c * K + k] = f2bf(W[idx]);
}

// ---------------------------------------------------------------------------
// Kernel 0b: convert x fp32 -> bf16 once (row-major), so the QKV GEMM's
// A-operand needs no per-iteration conversion (x is reused by 192 col tiles).
// ---------------------------------------------------------------------------
__global__ void cvt_x_kernel(const float* __restrict__ x,
                             bf16* __restrict__ xbf, int n8) {
    int i = blockIdx.x * blockDim.x + threadIdx.x;
    if (i >= n8) return;
    v8f f = *reinterpret_cast<const v8f*>(x + (size_t)i * 8);
    *reinterpret_cast<v8bf*>(xbf + (size_t)i * 8) = cvt8(f);
}

// ---------------------------------------------------------------------------
// Kernel 1: fused QKV projection + RoPE.
// One wave computes a 16x16 tile of [M=B*N rows] x [3072 cols] where cols
// 0..1023 -> q, 1024..2047 -> k, 2048..3071 -> v.  32 WMMAs per tile (K=1024).
// q/k get rotary (pair exchange = shfl_xor(1) since lanes hold columns),
// q gets the 1/sqrt(d) scale folded in.  q,k stored [b,h,n,d] bf16;
// v stored transposed [b,h,d,n] bf16 for contiguous PV B-operands.
// ---------------------------------------------------------------------------
__global__ __launch_bounds__(32)
void qkv_rope_kernel(const bf16* __restrict__ xbf,
                     const float* __restrict__ pos_emb,
                     const bf16* __restrict__ WqT,
                     const bf16* __restrict__ WkvT,
                     bf16* __restrict__ qbuf,
                     bf16* __restrict__ kbuf,
                     bf16* __restrict__ vTbuf) {
    const int lane  = threadIdx.x & 31;
    const int half  = lane >> 4;
    const int mlane = lane & 15;
    const int r0 = blockIdx.x * 16;   // global row in [0, B*N)
    const int c0 = blockIdx.y * 16;   // global col in [0, 3*INNER)

    const bf16* WT;
    int wcol;
    if (c0 < INNER) { WT = WqT;  wcol = c0; }
    else            { WT = WkvT; wcol = c0 - INNER; }

    const bf16* xrow  = xbf + (size_t)(r0 + mlane) * DMODEL;    // A: lane's row
    const bf16* wcolp = WT + (size_t)(wcol + mlane) * DMODEL;   // B: lane's col

    v8f acc = {};
    #pragma unroll 4
    for (int kb = 0; kb < DMODEL; kb += 32) {
        prefetch_near(wcolp + kb + 256);             // stream weight column ahead
        acc = wmma_bf16(loadA_bf16(xrow, kb, half),
                        loadB_bf16(wcolp, kb, half), acc);
    }

    const int bidx  = r0 / SEQ;
    const int nbase = r0 % SEQ;
    const int cg    = c0 + mlane;

    if (c0 < 2 * INNER) {
        // q or k: apply rotary, scatter to [b,h,n,d]
        const int  cc  = (c0 < INNER) ? cg : (cg - INNER);
        const int  h   = cc >> 6, d = cc & 63;
        bf16*      dst = (c0 < INNER) ? qbuf : kbuf;
        const float sgn    = (d & 1) ? 1.0f : -1.0f;       // rotate_every_two sign
        const float qscale = (c0 < INNER) ? 0.125f : 1.0f; // DHEAD^-0.5 folded into q
        #pragma unroll
        for (int r = 0; r < 8; ++r) {
            int   n    = nbase + r + 8 * half;
            float val  = acc[r];
            float part = __shfl_xor(val, 1, 32);           // value at column d^1
            float pe   = pos_emb[n * DHEAD + d];
            float res  = (val * __cosf(pe) + sgn * part * __sinf(pe)) * qscale;
            dst[(((size_t)(bidx * NHEAD + h) * SEQ + n) << 6) + d] = f2bf(res);
        }
    } else {
        // v: store transposed [b,h,d,n]; 8 consecutive n per lane -> one 16B store
        const int cc = cg - 2 * INNER;
        const int h  = cc >> 6, d = cc & 63;
        v8bf pv;
        #pragma unroll
        for (int r = 0; r < 8; ++r) pv[r] = f2bf(acc[r]);
        *reinterpret_cast<v8bf*>(
            vTbuf + ((size_t)(bidx * NHEAD + h) * DHEAD + d) * SEQ
                  + nbase + 8 * half) = pv;
    }
}

// ---------------------------------------------------------------------------
// Kernel 2: flash attention.  One wave owns 16 query rows of one (b,h).
// K/V 32-key blocks are staged into LDS with async global->LDS copies and
// double buffering (issue next block, then s_wait_asynccnt 16 for current).
// Per block: 4 WMMAs QK^T (K=64), online softmax with shfl_xor row reductions,
// P re-laid-out C->A format through LDS (wave-local), 4 WMMAs PV.
// ---------------------------------------------------------------------------

// Stage one 32-key block: K rows [jb..jb+32) x 64 (contiguous 4KB) and
// vT rows 64 (d) x 32 (n) (64B per row, global row stride SEQ).
// 16 async b128 issues per call (32 lanes x 16B each).
static __device__ inline void stage_kv(const bf16* kbase, const bf16* vbase,
                                       int jb, bf16* kst, bf16* vst, int lane) {
    const bf16* ksrc = kbase + (size_t)jb * DHEAD;
    #pragma unroll
    for (int i = 0; i < 8; ++i) {
        int off = (i * 32 + lane) * 8;            // 16B chunks, linear
        cp16_async(kst + off, ksrc + off);
    }
    #pragma unroll
    for (int i = 0; i < 8; ++i) {
        int c = i * 32 + lane;
        int d = c >> 2, part = (c & 3) * 8;       // 4 chunks per d-row
        cp16_async(vst + d * 32 + part, vbase + (size_t)d * SEQ + jb + part);
    }
}

__global__ __launch_bounds__(32)
void flash_attn_kernel(const bf16* __restrict__ qbuf,
                       const bf16* __restrict__ kbuf,
                       const bf16* __restrict__ vTbuf,
                       const unsigned char* __restrict__ mask,  // numpy bool_: 1 byte
                       bf16* __restrict__ outH) {
    __shared__ alignas(64) bf16 kstage[2][32 * DHEAD];   // 2 x 4KB
    __shared__ alignas(64) bf16 vstage[2][DHEAD * 32];   // 2 x 4KB
    __shared__ alignas(32) float p_lds[16 * 32];         // 2KB

    const int lane  = threadIdx.x & 31;
    const int half  = lane >> 4;
    const int mlane = lane & 15;
    const int i0 = blockIdx.x * 16;
    const int h  = blockIdx.y;
    const int b  = blockIdx.z;
    const size_t bh = (size_t)b * NHEAD + h;

    const unsigned char* mrow = mask + (size_t)b * SEQ;
    const bf16* kbase = kbuf  + bh * SEQ * DHEAD;
    const bf16* vbase = vTbuf + bh * DHEAD * SEQ;

    // Kick off the first K/V block before loading q (overlap latency).
    stage_kv(kbase, vbase, 0, kstage[0], vstage[0], lane);

    // Preload this wave's 16 q rows (full K=64) as two A operands.
    const bf16* qrow = qbuf + (bh * SEQ + i0 + mlane) * DHEAD;
    const v16bf aq0 = loadA_bf16(qrow, 0, half);
    const v16bf aq1 = loadA_bf16(qrow, 32, half);

    v8f o0 = {}, o1 = {}, o2 = {}, o3 = {};
    float mrun[8], srun[8];
    #pragma unroll
    for (int r = 0; r < 8; ++r) { mrun[r] = -3.0e38f; srun[r] = 0.0f; }

    int buf = 0;
    for (int jb = 0; jb < SEQ; jb += 32) {
        // Issue next block into the other buffer, then wait for the current
        // block (ASYNCcnt <= 16: in-order completion of the older 16 loads).
        if (jb + 32 < SEQ) {
            stage_kv(kbase, vbase, jb + 32, kstage[buf ^ 1], vstage[buf ^ 1], lane);
            async_wait16();
        } else {
            async_wait0();
        }
        const bf16* kst = kstage[buf];
        const bf16* vst = vstage[buf];

        // ---- QK^T from LDS: two 16x16 sim tiles, K=64 each ----
        const bf16* kcol0 = kst + (size_t)mlane * DHEAD;         // rows 0..15
        const bf16* kcol1 = kst + (size_t)(16 + mlane) * DHEAD;  // rows 16..31
        v8f c0t = {}, c1t = {};
        c0t = wmma_bf16(aq0, loadB_bf16(kcol0, 0,  half), c0t);
        c0t = wmma_bf16(aq1, loadB_bf16(kcol0, 32, half), c0t);
        c1t = wmma_bf16(aq0, loadB_bf16(kcol1, 0,  half), c1t);
        c1t = wmma_bf16(aq1, loadB_bf16(kcol1, 32, half), c1t);

        const unsigned char mk0 = mrow[jb + mlane];
        const unsigned char mk1 = mrow[jb + 16 + mlane];

        // ---- online softmax over the joint 32-key block ----
        #pragma unroll
        for (int r = 0; r < 8; ++r) {
            float v0 = mk0 ? c0t[r] : -3.0e38f;
            float v1 = mk1 ? c1t[r] : -3.0e38f;
            float mx   = fmaxf(rowmax16(v0), rowmax16(v1));
            float mnew = fmaxf(mrun[r], mx);
            float corr = __expf(mrun[r] - mnew);
            float p0   = __expf(v0 - mnew);
            float p1   = __expf(v1 - mnew);
            srun[r] = srun[r] * corr + rowsum16(p0) + rowsum16(p1);
            mrun[r] = mnew;
            o0[r] *= corr; o1[r] *= corr; o2[r] *= corr; o3[r] *= corr;
            const int prow = r + 8 * half;           // C-layout row of this elem
            p_lds[prow * 32 + mlane]      = p0;
            p_lds[prow * 32 + 16 + mlane] = p1;
        }

        // ---- P: C-layout -> A-layout via LDS (same wave; HW waits on DScnt) ----
        const float* pr = p_lds + mlane * 32;
        v8f g0 = *reinterpret_cast<const v8f*>(pr + 16 * half);
        v8f g1 = *reinterpret_cast<const v8f*>(pr + 16 * half + 8);
        v16bf ap = cat8(cvt8(g0), cvt8(g1));

        // ---- PV from LDS: out[16x64] += P[16x32] @ V[32x64] ----
        const bf16* vcol = vst + (size_t)mlane * 32 + 16 * half;
        o0 = wmma_bf16(ap, *reinterpret_cast<const v16bf*>(vcol + 0 * 16 * 32), o0);
        o1 = wmma_bf16(ap, *reinterpret_cast<const v16bf*>(vcol + 1 * 16 * 32), o1);
        o2 = wmma_bf16(ap, *reinterpret_cast<const v16bf*>(vcol + 2 * 16 * 32), o2);
        o3 = wmma_bf16(ap, *reinterpret_cast<const v16bf*>(vcol + 3 * 16 * 32), o3);

        buf ^= 1;
    }

    // ---- normalize and store per-head output [b, n, h*64+d] bf16 ----
    #pragma unroll
    for (int r = 0; r < 8; ++r) {
        int n = i0 + r + 8 * half;
        float inv = 1.0f / srun[r];
        size_t base = ((size_t)b * SEQ + n) * INNER + h * DHEAD + mlane;
        outH[base +  0] = f2bf(o0[r] * inv);
        outH[base + 16] = f2bf(o1[r] * inv);
        outH[base + 32] = f2bf(o2[r] * inv);
        outH[base + 48] = f2bf(o3[r] * inv);
    }
}

// ---------------------------------------------------------------------------
// Kernel 3: output projection outH[B*N, INNER] @ Wout[INNER, DMODEL] + bout.
// ---------------------------------------------------------------------------
__global__ __launch_bounds__(32)
void outproj_kernel(const bf16* __restrict__ outH,
                    const bf16* __restrict__ WoutT,
                    const float* __restrict__ bout,
                    float* __restrict__ out) {
    const int lane  = threadIdx.x & 31;
    const int half  = lane >> 4;
    const int mlane = lane & 15;
    const int r0 = blockIdx.x * 16;
    const int c0 = blockIdx.y * 16;

    const bf16* arow  = outH  + (size_t)(r0 + mlane) * INNER;
    const bf16* wcolp = WoutT + (size_t)(c0 + mlane) * INNER;

    v8f acc = {};
    #pragma unroll 4
    for (int kb = 0; kb < INNER; kb += 32) {
        prefetch_near(wcolp + kb + 256);
        acc = wmma_bf16(loadA_bf16(arow, kb, half),
                        loadB_bf16(wcolp, kb, half), acc);
    }

    const float bias = bout[c0 + mlane];
    #pragma unroll
    for (int r = 0; r < 8; ++r) {
        int row = r0 + r + 8 * half;
        out[(size_t)row * DMODEL + c0 + mlane] = acc[r] + bias;
    }
}

// ---------------------------------------------------------------------------
// Workspace layout (bytes):
//   WqT   bf16 [INNER,  DMODEL] : 2 MB
//   WkvT  bf16 [2*INNER,DMODEL] : 4 MB
//   WoutT bf16 [DMODEL, INNER ] : 2 MB
//   xbf   bf16 [B*N, DMODEL]    : 8 MB
//   q     bf16 [B,H,N,D]        : 8 MB
//   k     bf16 [B,H,N,D]        : 8 MB
//   vT    bf16 [B,H,D,N]        : 8 MB
//   outH  bf16 [B,N,INNER]      : 8 MB      (total ~48 MB)
// ---------------------------------------------------------------------------
extern "C" void kernel_launch(void* const* d_in, const int* in_sizes, int n_in,
                              void* d_out, int out_size, void* d_ws, size_t ws_size,
                              hipStream_t stream) {
    (void)in_sizes; (void)n_in; (void)out_size; (void)ws_size;

    const float* x       = (const float*)d_in[0];
    const unsigned char* mask = (const unsigned char*)d_in[1];  // jnp bool -> 1 byte
    const float* pos_emb = (const float*)d_in[2];
    const float* Wq      = (const float*)d_in[3];
    const float* Wkv     = (const float*)d_in[4];
    const float* Wout    = (const float*)d_in[5];
    const float* bout    = (const float*)d_in[6];
    float* out           = (float*)d_out;

    char* ws = (char*)d_ws;
    const size_t SZ_WQT  = (size_t)INNER * DMODEL * 2;
    const size_t SZ_WKVT = (size_t)2 * INNER * DMODEL * 2;
    const size_t SZ_X    = (size_t)BATCH * SEQ * DMODEL * 2;
    const size_t SZ_HEAD = (size_t)BATCH * NHEAD * SEQ * DHEAD * 2;
    bf16* WqT   = (bf16*)(ws);
    bf16* WkvT  = (bf16*)(ws + SZ_WQT);
    bf16* WoutT = (bf16*)(ws + SZ_WQT + SZ_WKVT);
    bf16* xbf   = (bf16*)(ws + SZ_WQT + SZ_WKVT + SZ_WQT);
    bf16* qbuf  = (bf16*)((char*)xbf + SZ_X);
    bf16* kbuf  = (bf16*)((char*)qbuf + SZ_HEAD);
    bf16* vTbuf = (bf16*)((char*)kbuf + SZ_HEAD);
    bf16* outH  = (bf16*)((char*)vTbuf + SZ_HEAD);

    // 0) one-time conversions (negligible vs GEMMs)
    {
        int n1 = DMODEL * INNER;
        transpose_cvt_kernel<<<(n1 + 255) / 256, 256, 0, stream>>>(Wq, WqT, DMODEL, INNER);
        int n2 = DMODEL * 2 * INNER;
        transpose_cvt_kernel<<<(n2 + 255) / 256, 256, 0, stream>>>(Wkv, WkvT, DMODEL, 2 * INNER);
        int n3 = INNER * DMODEL;
        transpose_cvt_kernel<<<(n3 + 255) / 256, 256, 0, stream>>>(Wout, WoutT, INNER, DMODEL);
        int n8 = BATCH * SEQ * DMODEL / 8;
        cvt_x_kernel<<<(n8 + 255) / 256, 256, 0, stream>>>(x, xbf, n8);
    }

    // 1) fused QKV GEMM + RoPE: (B*N/16) x (3*INNER/16) wave tiles
    {
        dim3 grid(BATCH * SEQ / 16, 3 * INNER / 16);
        qkv_rope_kernel<<<grid, 32, 0, stream>>>(xbf, pos_emb, WqT, WkvT,
                                                 qbuf, kbuf, vTbuf);
    }

    // 2) flash attention: (N/16, H, B) wave tiles
    {
        dim3 grid(SEQ / 16, NHEAD, BATCH);
        flash_attn_kernel<<<grid, 32, 0, stream>>>(qbuf, kbuf, vTbuf, mask, outH);
    }

    // 3) output projection + bias
    {
        dim3 grid(BATCH * SEQ / 16, DMODEL / 16);
        outproj_kernel<<<grid, 32, 0, stream>>>(outH, WoutT, bout, out);
    }
}
